// TARGET_LSTM_39101382263301
// MI455X (gfx1250) — compile-verified, matrix operand order checked by reading, not run on hardware
//
#include <hip/hip_runtime.h>
#include <hip/hip_bf16.h>
#include <stdint.h>

// ---------------------------------------------------------------------------
// MI455X (gfx1250) LSTM LM forward + NLL loss.
//  - bf16 WMMA (v_wmma_f32_16x16x32_bf16) for gate GEMMs and the dominant
//    [4096,256] x [256,32000] output projection (67 GFLOP, compute bound;
//    Wo(bf16)=16.4MB lives in the 192MB L2 after one HBM pass).
//  - f32 accumulate, f32 cell state / activations, exact f32 target logits.
//  - Raw v_exp_f32/v_log_f32 transcendentals (branch-free) keep the VALU
//    stream thin so it co-executes with the XDL WMMA pipe.
//  - Recurrence K-loops NOT unrolled (8 waves/SIMD -> ~128 VGPR budget);
//    all 4 gates' B-fragments staged in distinct regs so one s_wait covers
//    an 8-load clause followed by 4 back-to-back WMMAs.
// ---------------------------------------------------------------------------

typedef __attribute__((ext_vector_type(16))) __bf16 v16bf;
typedef __attribute__((ext_vector_type(8)))  float  v8f;

#if __has_builtin(__builtin_amdgcn_exp2f)
#define EXP2F(x) __builtin_amdgcn_exp2f(x)
#else
#define EXP2F(x) exp2f(x)
#endif
#if __has_builtin(__builtin_amdgcn_logf)
#define LOG2F(x) __builtin_amdgcn_logf(x)
#else
#define LOG2F(x) log2f(x)
#endif
#if __has_builtin(__builtin_amdgcn_rcpf)
#define RCPF(x) __builtin_amdgcn_rcpf(x)
#else
#define RCPF(x) (1.0f / (x))
#endif

#define LOG2E 1.4426950408889634f
#define LN2   0.6931471805599453f

namespace {
constexpr int Vv   = 32000;
constexpr int Bb   = 32;
constexpr int Ee   = 64;
constexpr int Hh   = 256;
constexpr int Tt   = 128;
constexpr int ROWS = Tt * Bb;          // 4096
constexpr int NTV  = Vv / 16;          // 2000 N-tiles over vocab
constexpr int NCHUNK = 8;              // vocab chunks for extra WG parallelism
constexpr int NT_PER_CHUNK = NTV / NCHUNK;   // 250

// workspace layout (bytes, 256B aligned)
constexpr size_t WCAT_OFF = 0;                          // 640 tiles *512 bf16 = 655360 B
constexpr size_t BCAT_OFF = WCAT_OFF + 655360;          // 1024 f32 = 4096 B
constexpr size_t WO_OFF   = BCAT_OFF + 4096;            // 16000 tiles *512 bf16 = 16384000 B
constexpr size_t HS_OFF   = WO_OFF + 16384000;          // 4096*256 bf16 = 2097152 B
constexpr size_t MAXP_OFF = HS_OFF + 2097152;           // 8*4096 f32 = 131072 B
constexpr size_t SUMP_OFF = MAXP_OFF + 131072;          // 131072 B
constexpr size_t MAXV_OFF = SUMP_OFF + 131072;          // 16384 B
constexpr size_t SUMV_OFF = MAXV_OFF + 16384;           // 16384 B
constexpr size_t TGT_OFF  = SUMV_OFF + 16384;           // 16384 B
} // namespace

union BF16Frag {
  v16bf v;
  unsigned short us[16];
  unsigned int   u32[8];
  uint4          q[2];
};

__device__ __forceinline__ unsigned short f2bf(float f) {
  unsigned int u = __float_as_uint(f);
  unsigned int r = u + 0x7FFFu + ((u >> 16) & 1u);   // round-to-nearest-even
  return (unsigned short)(r >> 16);
}
__device__ __forceinline__ float bf2f(unsigned short h) {
  return __uint_as_float(((unsigned int)h) << 16);
}
// branch-free sigmoid / tanh on the raw exp2 pipe (inputs here are O(1))
__device__ __forceinline__ float fsigmoid(float x) {
  return RCPF(1.f + EXP2F(-x * LOG2E));
}
__device__ __forceinline__ float ftanh(float x) {
  return 1.f - 2.f * RCPF(1.f + EXP2F(2.f * LOG2E * x));
}
// K-local index of bf16 A/B fragment element pair v (dword v) for half hf,
// per CDNA5 16-bit A-matrix 16x32 VGPR layout (CDNA5 ISA 7.12.2).
__device__ __forceinline__ int klocal_pair(int v, int hf) {
  return (v < 4) ? (2 * v + 8 * hf) : (16 + 2 * (v - 4) + 8 * hf);
}

// ---------------------------------------------------------------------------
// Pack concatenated gate weights [K=320][N=1024] (gates i,f,o,c along N;
// rows 0..63 = W*, rows 64..319 = U*) into bf16 fragment tiles:
// tile = nt*10 + kt (nt: 64 N-tiles, kt: 10 K-tiles of 32), 32 lanes * 16 elem.
// ---------------------------------------------------------------------------
__global__ __launch_bounds__(256) void k_pack_wcat(
    const float* __restrict__ Wi, const float* __restrict__ Ui,
    const float* __restrict__ Wf, const float* __restrict__ Uf,
    const float* __restrict__ Wog, const float* __restrict__ Uog,
    const float* __restrict__ Wc, const float* __restrict__ Uc,
    unsigned short* __restrict__ out) {
  int idx = blockIdx.x * 256 + threadIdx.x;          // < 640*512
  int e    = idx & 15;
  int lane = (idx >> 4) & 31;
  int tile = idx >> 9;
  int kt = tile % 10, nt = tile / 10;
  int nl = lane & 15, hf = lane >> 4;
  int kl = klocal_pair(e >> 1, hf) + (e & 1);
  int k  = kt * 32 + kl;                              // 0..319
  int j  = nt * 16 + nl;                              // 0..1023
  int g  = j >> 8, jj = j & 255;
  const float* W = (g == 0) ? Wi : (g == 1) ? Wf : (g == 2) ? Wog : Wc;
  const float* U = (g == 0) ? Ui : (g == 1) ? Uf : (g == 2) ? Uog : Uc;
  float src = (k < Ee) ? W[k * Hh + jj] : U[(k - Ee) * Hh + jj];
  out[idx] = f2bf(src);
}

__global__ __launch_bounds__(256) void k_pack_bias(
    const float* __restrict__ bi, const float* __restrict__ bf_,
    const float* __restrict__ bog, const float* __restrict__ bc,
    float* __restrict__ bcat) {
  int j = blockIdx.x * 256 + threadIdx.x;            // < 1024
  int g = j >> 8, jj = j & 255;
  const float* b = (g == 0) ? bi : (g == 1) ? bf_ : (g == 2) ? bog : bc;
  bcat[j] = b[jj];
}

// Pack Wo [K=256][V=32000] into bf16 fragment tiles: tile = nt*8 + kt.
__global__ __launch_bounds__(256) void k_pack_wo(
    const float* __restrict__ Wo, unsigned short* __restrict__ out) {
  size_t idx = (size_t)blockIdx.x * 256 + threadIdx.x;  // < 16000*512
  int e    = (int)(idx & 15);
  int lane = (int)((idx >> 4) & 31);
  size_t tile = idx >> 9;
  int kt = (int)(tile & 7);
  size_t nt = tile >> 3;
  int nl = lane & 15, hf = lane >> 4;
  int kl = klocal_pair(e >> 1, hf) + (e & 1);
  int k  = kt * 32 + kl;                               // 0..255
  size_t n = nt * 16 + nl;                             // 0..31999
  out[idx] = f2bf(Wo[(size_t)k * Vv + n]);
}

// ---------------------------------------------------------------------------
// Sequential LSTM recurrence. One persistent workgroup (32 waves, 8/SIMD).
// Wave w owns output tile (mt=w/16, ht=w%16) for all 4 gates; cell state
// stays in a v8f register fragment across all 128 steps. h double-buffered
// in LDS (re-read as A operand next step), and streamed to Hs (bf16).
// ---------------------------------------------------------------------------
__global__ __launch_bounds__(1024) void k_lstm_recurrence(
    const int* __restrict__ input_x, const float* __restrict__ emb,
    const unsigned short* __restrict__ wcat, const float* __restrict__ bcat,
    const int* __restrict__ start_tok, unsigned short* __restrict__ Hs) {
  __shared__ unsigned short xbuf[Bb * Ee];           // 4 KB
  __shared__ unsigned short hbuf[2][Bb * Hh];        // 32 KB
  const int tid  = threadIdx.x;
  const int lane = tid & 31;
  const int wave = tid >> 5;                         // 0..31
  const int nl = lane & 15, hf = lane >> 4;
  const int mt = wave >> 4;                          // 0..1  (batch row tile)
  const int ht = wave & 15;                          // 0..15 (hidden col tile)
  const int m  = mt * 16 + nl;                       // A-fragment row

  for (int i = tid; i < Bb * Hh; i += 1024) hbuf[0][i] = 0;

  v8f cst;
  #pragma unroll
  for (int r = 0; r < 8; ++r) cst[r] = 0.f;
  const int start = start_tok[0];
  // per-wave base of this wave's 40 weight tiles: tile = (g*16+ht)*10 + kt
  const unsigned short* wbase = wcat + (size_t)ht * 10 * 512 + (size_t)lane * 16;
  const float bv0 = bcat[0 * Hh + ht * 16 + nl];
  const float bv1 = bcat[1 * Hh + ht * 16 + nl];
  const float bv2 = bcat[2 * Hh + ht * 16 + nl];
  const float bv3 = bcat[3 * Hh + ht * 16 + nl];

  for (int t = 0; t < Tt; ++t) {
    __syncthreads();                                  // prior step's reads done
    // teacher-forcing input x_t -> LDS as bf16 (32x64)
    for (int i = tid; i < Bb * Ee; i += 1024) {
      int b = i >> 6, e = i & 63;
      int tok = (t == 0) ? start : input_x[b * Tt + (t - 1)];
      xbuf[i] = f2bf(emb[tok * Ee + e]);
    }
    __syncthreads();
    const unsigned short* hin = hbuf[t & 1];
    unsigned short* hout = hbuf[(t + 1) & 1];

    v8f acc[4];
    #pragma unroll
    for (int r = 0; r < 8; ++r) {
      acc[0][r] = bv0; acc[1][r] = bv1; acc[2][r] = bv2; acc[3][r] = bv3;
    }
    // K loop: kt 0..1 reads A from xbuf, kt 2..9 from hbuf.  unroll 1 keeps
    // the live set (~1 A frag + 4 staged B frags + 4 acc) within the
    // 8-waves/SIMD VGPR budget -> no scratch spills.
    #pragma unroll 1
    for (int kt = 0; kt < 10; ++kt) {
      // stage all 4 gates' B fragments: one 8x global_load_b128 clause
      BF16Frag b0, b1, b2, b3;
      {
        const uint4* p0 = (const uint4*)(wbase + ((size_t)0 * 160 + kt) * 512);
        const uint4* p1 = (const uint4*)(wbase + ((size_t)1 * 160 + kt) * 512);
        const uint4* p2 = (const uint4*)(wbase + ((size_t)2 * 160 + kt) * 512);
        const uint4* p3 = (const uint4*)(wbase + ((size_t)3 * 160 + kt) * 512);
        b0.q[0] = p0[0]; b0.q[1] = p0[1];
        b1.q[0] = p1[0]; b1.q[1] = p1[1];
        b2.q[0] = p2[0]; b2.q[1] = p2[1];
        b3.q[0] = p3[0]; b3.q[1] = p3[1];
      }
      BF16Frag a;
      if (kt < 2) {                                   // uniform branch
        #pragma unroll
        for (int v = 0; v < 8; ++v) {
          int k = kt * 32 + klocal_pair(v, hf);
          a.u32[v] = *(const unsigned int*)&xbuf[m * Ee + k];
        }
      } else {
        #pragma unroll
        for (int v = 0; v < 8; ++v) {
          int k = (kt - 2) * 32 + klocal_pair(v, hf);
          a.u32[v] = *(const unsigned int*)&hin[m * Hh + k];
        }
      }
      acc[0] = __builtin_amdgcn_wmma_f32_16x16x32_bf16(
          false, a.v, false, b0.v, (short)0, acc[0], false, false);
      acc[1] = __builtin_amdgcn_wmma_f32_16x16x32_bf16(
          false, a.v, false, b1.v, (short)0, acc[1], false, false);
      acc[2] = __builtin_amdgcn_wmma_f32_16x16x32_bf16(
          false, a.v, false, b2.v, (short)0, acc[2], false, false);
      acc[3] = __builtin_amdgcn_wmma_f32_16x16x32_bf16(
          false, a.v, false, b3.v, (short)0, acc[3], false, false);
    }
    // gate nonlinearities + cell/hidden update (rows per C/D layout)
    #pragma unroll
    for (int r = 0; r < 8; ++r) {
      float ig = fsigmoid(acc[0][r]);
      float fg = fsigmoid(acc[1][r]);
      float og = fsigmoid(acc[2][r]);
      float cb = ftanh(acc[3][r]);
      float c  = fg * cst[r] + ig * cb;
      cst[r] = c;
      float h = og * ftanh(c);
      int mrow = mt * 16 + r + 8 * hf;
      int j    = ht * 16 + nl;
      unsigned short hb = f2bf(h);
      hout[mrow * Hh + j] = hb;
      Hs[((size_t)t * Bb + mrow) * Hh + j] = hb;
    }
  }
}

// ---------------------------------------------------------------------------
// Fused logits GEMM + streaming log-sum-exp. blockIdx: chunk = b>>5 (vocab
// chunk of 250 N-tiles), mgroup = b&31. Each wave owns one 16-row M-tile
// exclusively; A fragments loaded once; per-lane branch-free online
// max/sumexp in exp2 domain (raw v_exp_f32, no divergence in the hot loop),
// cross-lane combine only at the end.
// ---------------------------------------------------------------------------
__global__ __launch_bounds__(256) void k_logits_softmax(
    const unsigned short* __restrict__ Hs, const unsigned short* __restrict__ wopk,
    const float* __restrict__ bo, float* __restrict__ maxp, float* __restrict__ sump) {
  const int tid  = threadIdx.x;
  const int lane = tid & 31;
  const int wave = tid >> 5;
  const int chunk  = blockIdx.x >> 5;                // 0..7
  const int mgroup = blockIdx.x & 31;                // 0..31
  const int mtile  = mgroup * 8 + wave;              // 0..255
  const int nl = lane & 15, hf = lane >> 4;
  const int m  = mtile * 16 + nl;

  BF16Frag a[8];                                     // K=256 -> 8 fragments
  #pragma unroll
  for (int kt = 0; kt < 8; ++kt)
    #pragma unroll
    for (int v = 0; v < 8; ++v) {
      int k = kt * 32 + klocal_pair(v, hf);
      a[kt].u32[v] = *(const unsigned int*)&Hs[(size_t)m * Hh + k];
    }

  float lmax[8], lsum[8];
  #pragma unroll
  for (int r = 0; r < 8; ++r) { lmax[r] = -1.0e30f; lsum[r] = 0.f; }

  const int nt0 = chunk * NT_PER_CHUNK;
  for (int it = 0; it < NT_PER_CHUNK; ++it) {
    const int nt = nt0 + it;
    v8f acc;
    #pragma unroll
    for (int r = 0; r < 8; ++r) acc[r] = 0.f;
    #pragma unroll
    for (int kt = 0; kt < 8; ++kt) {
      const uint4* bp = (const uint4*)(wopk +
          ((size_t)(nt * 8 + kt) * 32 + lane) * 16);
      BF16Frag bf; bf.q[0] = bp[0]; bf.q[1] = bp[1];
      acc = __builtin_amdgcn_wmma_f32_16x16x32_bf16(
          false, a[kt].v, false, bf.v, (short)0, acc, false, false);
    }
    float bv = bo[nt * 16 + nl];
    #pragma unroll
    for (int r = 0; r < 8; ++r) {
      float val = acc[r] + bv;
      float nm = fmaxf(lmax[r], val);
      lsum[r] = lsum[r] * EXP2F((lmax[r] - nm) * LOG2E)
              + EXP2F((val - nm) * LOG2E);
      lmax[r] = nm;
    }
  }
  // combine across the 16 lanes of each half-wave, then lane nl==0 writes.
  #pragma unroll
  for (int r = 0; r < 8; ++r) {
    float m0 = lmax[r], s0 = lsum[r];
    for (int w = 8; w >= 1; w >>= 1) {
      float m1 = __shfl_xor(m0, w, 16);
      float s1 = __shfl_xor(s0, w, 16);
      float nm = fmaxf(m0, m1);
      s0 = s0 * EXP2F((m0 - nm) * LOG2E) + s1 * EXP2F((m1 - nm) * LOG2E);
      m0 = nm;
    }
    if (nl == 0) {
      int row = mtile * 16 + r + 8 * hf;
      maxp[(size_t)chunk * ROWS + row] = m0;
      sump[(size_t)chunk * ROWS + row] = s0;
    }
  }
}

__global__ __launch_bounds__(256) void k_softmax_combine(
    const float* __restrict__ maxp, const float* __restrict__ sump,
    float* __restrict__ maxv, float* __restrict__ sumv) {
  int r = blockIdx.x * 256 + threadIdx.x;
  if (r >= ROWS) return;
  float m = -1.0e30f, s = 0.f;
  #pragma unroll
  for (int c = 0; c < NCHUNK; ++c) {
    float m1 = maxp[(size_t)c * ROWS + r], s1 = sump[(size_t)c * ROWS + r];
    float nm = fmaxf(m, m1);
    s = s * EXP2F((m - nm) * LOG2E) + s1 * EXP2F((m1 - nm) * LOG2E);
    m = nm;
  }
  maxv[r] = m; sumv[r] = s;
}

// Exact f32 target logit per (t,b) row: h_r . Wo[:,tgt] + bo[tgt].
__global__ __launch_bounds__(256) void k_target_logits(
    const unsigned short* __restrict__ Hs, const float* __restrict__ Wo,
    const float* __restrict__ bo, const int* __restrict__ input_x,
    float* __restrict__ tgt) {
  int r = blockIdx.x * 256 + threadIdx.x;
  if (r >= ROWS) return;
  int t = r / Bb, b = r % Bb;
  int v = input_x[b * Tt + t];
  float acc = bo[v];
  for (int k = 0; k < Hh; ++k)
    acc += bf2f(Hs[(size_t)r * Hh + k]) * Wo[(size_t)k * Vv + v];
  tgt[r] = acc;
}

// loss = sum_r (max_r + ln(sumexp_r) - tgt_logit_r)
__global__ __launch_bounds__(256) void k_loss_reduce(
    const float* __restrict__ maxv, const float* __restrict__ sumv,
    const float* __restrict__ tgt, float* __restrict__ out) {
  __shared__ float sh[256];
  float acc = 0.f;
  for (int r = threadIdx.x; r < ROWS; r += 256)
    acc += maxv[r] + LN2 * LOG2F(sumv[r]) - tgt[r];
  sh[threadIdx.x] = acc;
  __syncthreads();
  for (int s = 128; s > 0; s >>= 1) {
    if ((int)threadIdx.x < s) sh[threadIdx.x] += sh[threadIdx.x + s];
    __syncthreads();
  }
  if (threadIdx.x == 0) out[0] = sh[0];
}

// ---------------------------------------------------------------------------
extern "C" void kernel_launch(void* const* d_in, const int* in_sizes, int n_in,
                              void* d_out, int out_size, void* d_ws, size_t ws_size,
                              hipStream_t stream) {
  (void)in_sizes; (void)n_in; (void)out_size; (void)ws_size;
  const int*   input_x = (const int*)d_in[0];
  const float* emb  = (const float*)d_in[1];
  const float* Wi   = (const float*)d_in[2];
  const float* Ui   = (const float*)d_in[3];
  const float* bi   = (const float*)d_in[4];
  const float* Wf   = (const float*)d_in[5];
  const float* Uf   = (const float*)d_in[6];
  const float* bf_  = (const float*)d_in[7];
  const float* Wog  = (const float*)d_in[8];
  const float* Uog  = (const float*)d_in[9];
  const float* bog  = (const float*)d_in[10];
  const float* Wc   = (const float*)d_in[11];
  const float* Uc   = (const float*)d_in[12];
  const float* bc   = (const float*)d_in[13];
  const float* Wo   = (const float*)d_in[14];
  const float* bo   = (const float*)d_in[15];
  const int*   stt  = (const int*)d_in[16];

  char* ws = (char*)d_ws;
  unsigned short* wcat = (unsigned short*)(ws + WCAT_OFF);
  float*          bcat = (float*)(ws + BCAT_OFF);
  unsigned short* wopk = (unsigned short*)(ws + WO_OFF);
  unsigned short* hs   = (unsigned short*)(ws + HS_OFF);
  float*          maxp = (float*)(ws + MAXP_OFF);
  float*          sump = (float*)(ws + SUMP_OFF);
  float*          maxv = (float*)(ws + MAXV_OFF);
  float*          sumv = (float*)(ws + SUMV_OFF);
  float*          tgt  = (float*)(ws + TGT_OFF);

  k_pack_wcat<<<(640 * 512) / 256, 256, 0, stream>>>(Wi, Ui, Wf, Uf, Wog, Uog, Wc, Uc, wcat);
  k_pack_bias<<<4, 256, 0, stream>>>(bi, bf_, bog, bc, bcat);
  k_pack_wo<<<(16000 * 512) / 256, 256, 0, stream>>>(Wo, wopk);
  k_lstm_recurrence<<<1, 1024, 0, stream>>>(input_x, emb, wcat, bcat, stt, hs);
  k_logits_softmax<<<256, 256, 0, stream>>>(hs, wopk, bo, maxp, sump);
  k_softmax_combine<<<(ROWS + 255) / 256, 256, 0, stream>>>(maxp, sump, maxv, sumv);
  k_target_logits<<<(ROWS + 255) / 256, 256, 0, stream>>>(hs, Wo, bo, input_x, tgt);
  k_loss_reduce<<<1, 256, 0, stream>>>(maxv, sumv, tgt, (float*)d_out);
}
